// ConvSelfAttention_32186484916991
// MI455X (gfx1250) — matrix operand order repaired and checked by reading, not compiled, not run on hardware
//
#include <hip/hip_runtime.h>

typedef __attribute__((ext_vector_type(16))) _Float16 v16h;
typedef __attribute__((ext_vector_type(8)))  _Float16 v8h;
typedef __attribute__((ext_vector_type(8)))  float    v8f;
typedef __attribute__((ext_vector_type(2)))  _Float16 h2;

#define NTOK  4096
#define DHEAD 64
#define CIN   128

#define CAT16(lo, hi) __builtin_shufflevector(lo, hi, 0,1,2,3,4,5,6,7,8,9,10,11,12,13,14,15)

static __device__ __forceinline__ v8f wmma_f16(v16h a, v16h b, v8f c) {
    return __builtin_amdgcn_wmma_f32_16x16x32_f16(false, a, false, b, (short)0, c, false, false);
}

#if defined(__has_builtin)
#if __has_builtin(__builtin_amdgcn_global_load_async_to_lds_b128)
#define HAVE_ASYNC_LDS 1
#endif
#endif

#ifdef HAVE_ASYNC_LDS
// types per compiler diagnostic: v4i = int __attribute__((vector_size(16)))
typedef int v4i __attribute__((__vector_size__(16)));
typedef __attribute__((address_space(1))) v4i* gv4i_p;
typedef __attribute__((address_space(3))) v4i* lv4i_p;
// async DMA: 16 bytes per lane, global -> LDS, tracked by ASYNCcnt
static __device__ __forceinline__ void async_cp16(const _Float16* g, _Float16* l) {
    _Float16* gnc = const_cast<_Float16*>(g);
    __builtin_amdgcn_global_load_async_to_lds_b128((gv4i_p)gnc, (lv4i_p)l, 0, 0);
}
#endif

// =====================================================================
// Kernel 0: convert weights to f16 (once per launch, 64 KB total).
// =====================================================================
__global__ __launch_bounds__(256) void cvt_weights_kernel(
    const float* __restrict__ wq, const float* __restrict__ wk,
    const float* __restrict__ wv, const float* __restrict__ wo,
    _Float16* __restrict__ Whq, _Float16* __restrict__ Whk,
    _Float16* __restrict__ Whv, _Float16* __restrict__ Who)
{
    int idx = blockIdx.x * 256 + threadIdx.x;       // 0..32767
    if (idx < 8192)               Whq[idx]         = (_Float16)wq[idx];
    else if (idx < 16384)         Whk[idx - 8192]  = (_Float16)wk[idx - 8192];
    else if (idx < 24576)         Whv[idx - 16384] = (_Float16)wv[idx - 16384];
    else                          Who[idx - 24576] = (_Float16)wo[idx - 24576];
}

// =====================================================================
// Kernel 1: WMMA QKV projection. Block = 128-token tile, 8 waves.
//   C[16n x 16o] = Xs[16n x 32c] * W^T[32c x 16o], 4 k-steps, 3 projections.
//   Qt[b][n][o] (f16, (q+bq)*log2e/sqrt128), Kt[b][n][o], Vn[b][o][n]
// =====================================================================
__global__ __launch_bounds__(256) void qkv_proj_kernel(
    const float* __restrict__ x,
    const _Float16* __restrict__ Whq, const float* __restrict__ bq,
    const _Float16* __restrict__ Whk, const float* __restrict__ bk,
    const _Float16* __restrict__ Whv, const float* __restrict__ bv,
    _Float16* __restrict__ Qt, _Float16* __restrict__ Kt,
    _Float16* __restrict__ Vn)
{
    __shared__ _Float16 Xs[128][CIN];     // [n][c] f16, 32 KB
    __shared__ _Float16 Vt_s[DHEAD][128]; // [o][n] f16, 16 KB (V transpose buffer)

    const int b  = blockIdx.x >> 5;                // 32 tiles per batch
    const int n0 = (blockIdx.x & 31) * 128;
    const float* xb = x + (size_t)b * CIN * NTOK;

    // transpose-stage x tile: [c][n] f32 global -> [n][c] f16 LDS (pair-packed)
    #pragma unroll
    for (int it = 0; it < 32; ++it) {
        int idx = threadIdx.x + it * 256;          // 0..8191
        int c2 = idx >> 7;                         // 0..63 (pair of channels)
        int nl = idx & 127;
        h2 p;
        p.x = (_Float16)xb[(size_t)(2 * c2) * NTOK + n0 + nl];
        p.y = (_Float16)xb[(size_t)(2 * c2 + 1) * NTOK + n0 + nl];
        *(h2*)&Xs[nl][2 * c2] = p;
    }
    __syncthreads();

    const int wave = threadIdx.x >> 5;             // n-tile of 16 per wave
    const int lane = threadIdx.x & 31;
    const int m  = lane & 15;
    const int hh = lane >> 4;

    v8f acc[3][4] = {{{},{},{},{}},{{},{},{},{}},{{},{},{},{}}};  // [proj][o-tile]
    const _Float16* Wp[3] = { Whq, Whk, Whv };

    #pragma unroll
    for (int k = 0; k < 4; ++k) {
        // A-fragment of X: lane row n, elements K=c (two 16B chunks)
        v8h lo = *(const v8h*)&Xs[wave * 16 + m][k * 32 + 16 * hh];
        v8h hi = *(const v8h*)&Xs[wave * 16 + m][k * 32 + 8 + 16 * hh];
        v16h xa = CAT16(lo, hi);
        #pragma unroll
        for (int p = 0; p < 3; ++p) {
            #pragma unroll
            for (int ot = 0; ot < 4; ++ot) {
                // B-fragment: lane col o, elements K=c contiguous in W row
                v16h wb = *(const v16h*)&Wp[p][(size_t)(ot * 16 + m) * CIN + k * 32 + 16 * hh];
                acc[p][ot] = wmma_f16(xa, wb, acc[p][ot]);
            }
        }
    }

    // epilogue: bias, scale, f16 stores. C layout: VGPR r -> row n += r+8*hh
    const float qs = 0.12751649769930993f;         // log2(e)/sqrt(C_IN)
    #pragma unroll
    for (int ot = 0; ot < 4; ++ot) {
        int o = ot * 16 + m;
        float bqv = bq[o], bkv = bk[o], bvv = bv[o];
        #pragma unroll
        for (int r = 0; r < 8; ++r) {
            int n = n0 + wave * 16 + r + 8 * hh;
            Qt[((size_t)b * NTOK + n) * DHEAD + o] = (_Float16)((acc[0][ot][r] + bqv) * qs);
            Kt[((size_t)b * NTOK + n) * DHEAD + o] = (_Float16)(acc[1][ot][r] + bkv);
            Vt_s[o][wave * 16 + r + 8 * hh]        = (_Float16)(acc[2][ot][r] + bvv);
        }
    }
    __syncthreads();
    // coalesced V write-out: [o][n] LDS -> Vn[b][o][n] global, b128
    {
        int o = threadIdx.x >> 2, seg = (threadIdx.x & 3) * 32;
        #pragma unroll
        for (int u = 0; u < 4; ++u)
            *(uint4*)&Vn[((size_t)b * DHEAD + o) * NTOK + n0 + seg + u * 8] =
                *(const uint4*)&Vt_s[o][seg + u * 8];
    }
}

// =====================================================================
// Kernel 2: flash attention, WMMA f32_16x16x32_f16, 64-wide j-steps.
//   One wave = 16 query rows; block = 8 waves = 128 rows.
//   K/V tiles double-buffered in LDS with async global->LDS DMA (when
//   the toolchain exposes the gfx1250 async builtins).
//   AOh[b][n][o] f16 (softmax-normalized attention output)
// =====================================================================
__global__ __launch_bounds__(256) void flash_attn_kernel(
    const _Float16* __restrict__ Qt, const _Float16* __restrict__ Kt,
    const _Float16* __restrict__ Vn, _Float16* __restrict__ AOh)
{
#ifdef HAVE_ASYNC_LDS
    __shared__ _Float16 Ks[2 * 64][DHEAD];   // [buf*64 + j][o]  (16 KB)
    __shared__ _Float16 Vs[2 * DHEAD][64];   // [buf*64 + o][j]  (16 KB)
#else
    __shared__ _Float16 Ks[64][DHEAD];       // [j][o]   K tile (8 KB)
    __shared__ _Float16 Vs[DHEAD][64];       // [o][j]   V tile (8 KB)
#endif
    __shared__ _Float16 Ps[8][16][64];       // per-wave P tile [i][j] (16 KB)

    const int b    = blockIdx.x >> 5;        // 32 i-blocks per batch
    const int iblk = blockIdx.x & 31;
    const int wave = threadIdx.x >> 5;
    const int lane = threadIdx.x & 31;
    const int m  = lane & 15;
    const int hh = lane >> 4;
    const int i0 = iblk * 128 + wave * 16;

    const _Float16* Qb = Qt + (size_t)b * NTOK * DHEAD;
    const _Float16* Kb = Kt + (size_t)b * NTOK * DHEAD;
    const _Float16* Vb = Vn + (size_t)b * DHEAD * NTOK;

    // Q A-fragments resident in VGPRs (2x b128 each + concat)
    v16h qa0, qa1;
    {
        const _Float16* qr = &Qb[(size_t)(i0 + m) * DHEAD];
        v8h l0 = *(const v8h*)&qr[16 * hh];
        v8h h0 = *(const v8h*)&qr[8 + 16 * hh];
        v8h l1 = *(const v8h*)&qr[32 + 16 * hh];
        v8h h1 = *(const v8h*)&qr[40 + 16 * hh];
        qa0 = CAT16(l0, h0);
        qa1 = CAT16(l1, h1);
    }

    v8f oacc[4] = {{}, {}, {}, {}};          // O: 16 x 64 f32 accumulators
    float mrow[8], lrow[8];
    #pragma unroll
    for (int r = 0; r < 8; ++r) { mrow[r] = -1e30f; lrow[r] = 0.0f; }

    const int sr = threadIdx.x >> 2;             // staging row 0..63
    const int sc = (threadIdx.x & 3) * 16;       // staging col base (halves)

#ifdef HAVE_ASYNC_LDS
    // ---- prologue: async DMA of first K/V tile into buffer 0 ----
    async_cp16(&Kb[(size_t)sr * DHEAD + sc],     &Ks[sr][sc]);
    async_cp16(&Kb[(size_t)sr * DHEAD + sc + 8], &Ks[sr][sc + 8]);
    async_cp16(&Vb[(size_t)sr * NTOK + sc],      &Vs[sr][sc]);
    async_cp16(&Vb[(size_t)sr * NTOK + sc + 8],  &Vs[sr][sc + 8]);
    asm volatile("s_wait_asynccnt 0x0" ::: "memory");
    __syncthreads();
#endif

    for (int j0 = 0; j0 < NTOK; j0 += 64) {
#ifdef HAVE_ASYNC_LDS
        const int buf  = (j0 >> 6) & 1;
        const int nbuf = buf ^ 1;
        // issue next tile's DMA; overlaps with this tile's WMMAs + softmax
        if (j0 + 64 < NTOK) {
            const int jn = j0 + 64;
            async_cp16(&Kb[(size_t)(jn + sr) * DHEAD + sc],     &Ks[nbuf * 64 + sr][sc]);
            async_cp16(&Kb[(size_t)(jn + sr) * DHEAD + sc + 8], &Ks[nbuf * 64 + sr][sc + 8]);
            async_cp16(&Vb[(size_t)sr * NTOK + jn + sc],        &Vs[nbuf * 64 + sr][sc]);
            async_cp16(&Vb[(size_t)sr * NTOK + jn + sc + 8],    &Vs[nbuf * 64 + sr][sc + 8]);
        }
        const int kbase = buf * 64;
        const int vbase = buf * 64;
#else
        // ---- fallback: cooperative staging through VGPRs + prefetch ----
        *(uint4*)&Ks[sr][sc]     = *(const uint4*)&Kb[(size_t)(j0 + sr) * DHEAD + sc];
        *(uint4*)&Ks[sr][sc + 8] = *(const uint4*)&Kb[(size_t)(j0 + sr) * DHEAD + sc + 8];
        *(uint4*)&Vs[sr][sc]     = *(const uint4*)&Vb[(size_t)sr * NTOK + j0 + sc];
        *(uint4*)&Vs[sr][sc + 8] = *(const uint4*)&Vb[(size_t)sr * NTOK + j0 + sc + 8];
        if (j0 + 64 < NTOK) {
            __builtin_prefetch(&Kb[(size_t)(j0 + 64 + sr) * DHEAD + sc], 0, 0);
            __builtin_prefetch(&Vb[(size_t)sr * NTOK + j0 + 64 + sc], 0, 0);
        }
        __syncthreads();
        const int kbase = 0;
        const int vbase = 0;
#endif

        // ---- S = Q K^T over 64 j-columns (four 16x16 C fragments) ----
        v8f s[4];
        #pragma unroll
        for (int jq = 0; jq < 4; ++jq) {
            v16h kb0 = *(const v16h*)&Ks[kbase + jq * 16 + m][hh * 16];
            v16h kb1 = *(const v16h*)&Ks[kbase + jq * 16 + m][32 + hh * 16];
            v8f a = {};
            a = wmma_f16(qa0, kb0, a);
            a = wmma_f16(qa1, kb1, a);
            s[jq] = a;
        }

        // ---- online softmax (base-2); C layout: VGPR r -> row r+8*hh ----
        #pragma unroll
        for (int r = 0; r < 8; ++r) {
            float mx = fmaxf(fmaxf(s[0][r], s[1][r]), fmaxf(s[2][r], s[3][r]));
            mx = fmaxf(mx, __shfl_xor(mx, 1));
            mx = fmaxf(mx, __shfl_xor(mx, 2));
            mx = fmaxf(mx, __shfl_xor(mx, 4));
            mx = fmaxf(mx, __shfl_xor(mx, 8));
            float mnew = fmaxf(mrow[r], mx);
            float a  = __builtin_amdgcn_exp2f(mrow[r] - mnew);
            float p0 = __builtin_amdgcn_exp2f(s[0][r] - mnew);
            float p1 = __builtin_amdgcn_exp2f(s[1][r] - mnew);
            float p2 = __builtin_amdgcn_exp2f(s[2][r] - mnew);
            float p3 = __builtin_amdgcn_exp2f(s[3][r] - mnew);
            float rs = (p0 + p1) + (p2 + p3);
            rs += __shfl_xor(rs, 1);
            rs += __shfl_xor(rs, 2);
            rs += __shfl_xor(rs, 4);
            rs += __shfl_xor(rs, 8);
            lrow[r] = lrow[r] * a + rs;
            mrow[r] = mnew;
            _Float16* pr = &Ps[wave][r + 8 * hh][0];
            pr[m]      = (_Float16)p0;
            pr[16 + m] = (_Float16)p1;
            pr[32 + m] = (_Float16)p2;
            pr[48 + m] = (_Float16)p3;
            oacc[0][r] *= a; oacc[1][r] *= a; oacc[2][r] *= a; oacc[3][r] *= a;
        }

        // ---- P (C layout) -> two A fragments through LDS, 2x b128 each ----
        v16h pa0, pa1;
        {
            const _Float16* prr = &Ps[wave][m][0];
            v8h l0 = *(const v8h*)&prr[16 * hh];
            v8h h0 = *(const v8h*)&prr[8 + 16 * hh];
            v8h l1 = *(const v8h*)&prr[32 + 16 * hh];
            v8h h1 = *(const v8h*)&prr[40 + 16 * hh];
            pa0 = CAT16(l0, h0);
            pa1 = CAT16(l1, h1);
        }

        // ---- O += P * V  (4 o-chunks x 2 k-steps) ----
        #pragma unroll
        for (int oq = 0; oq < 4; ++oq) {
            v16h vb0 = *(const v16h*)&Vs[vbase + oq * 16 + m][hh * 16];
            v16h vb1 = *(const v16h*)&Vs[vbase + oq * 16 + m][32 + hh * 16];
            oacc[oq] = wmma_f16(pa0, vb0, oacc[oq]);
            oacc[oq] = wmma_f16(pa1, vb1, oacc[oq]);
        }

#ifdef HAVE_ASYNC_LDS
        asm volatile("s_wait_asynccnt 0x0" ::: "memory");
#endif
        __syncthreads();
    }

    // ---- normalize and write AOh[b][n][o] f16 ----
    #pragma unroll
    for (int r = 0; r < 8; ++r) {
        float inv = __builtin_amdgcn_rcpf(lrow[r]);
        size_t row = (size_t)b * NTOK + i0 + r + 8 * hh;
        AOh[row * DHEAD +  0 + m] = (_Float16)(oacc[0][r] * inv);
        AOh[row * DHEAD + 16 + m] = (_Float16)(oacc[1][r] * inv);
        AOh[row * DHEAD + 32 + m] = (_Float16)(oacc[2][r] * inv);
        AOh[row * DHEAD + 48 + m] = (_Float16)(oacc[3][r] * inv);
    }
}

// =====================================================================
// Kernel 3: WMMA output projection + gamma residual.
//   out[b][c][n] = gamma * (wo[c][:] . AO[b][n][:] + bo[c]) + x[b][c][n]
// =====================================================================
__global__ __launch_bounds__(256) void out_proj_kernel(
    const float* __restrict__ x, const _Float16* __restrict__ Who,
    const float* __restrict__ bo, const float* __restrict__ gamma,
    const _Float16* __restrict__ AOh, float* __restrict__ out)
{
    __shared__ _Float16 As[128][DHEAD];   // [n][o] f16, 16 KB
    const int b  = blockIdx.x >> 5;
    const int n0 = (blockIdx.x & 31) * 128;

    // direct b128 staging of AOh rows
    #pragma unroll
    for (int it = 0; it < 4; ++it) {
        int idx = threadIdx.x + it * 256;          // 0..1023
        int nl = idx >> 3, seg = (idx & 7) * 8;
        *(uint4*)&As[nl][seg] =
            *(const uint4*)&AOh[((size_t)b * NTOK + n0 + nl) * DHEAD + seg];
    }
    __syncthreads();

    const int wave = threadIdx.x >> 5;             // c-tile of 16 per wave
    const int lane = threadIdx.x & 31;
    const int m  = lane & 15;
    const int hh = lane >> 4;

    // A-fragments of wo (f16): lane row c, elements K=o, 2 k-steps
    v16h wa[2];
    #pragma unroll
    for (int k = 0; k < 2; ++k) {
        const _Float16* wr = &Who[(size_t)(wave * 16 + m) * DHEAD + k * 32];
        v8h lo = *(const v8h*)&wr[16 * hh];
        v8h hi = *(const v8h*)&wr[8 + 16 * hh];
        wa[k] = CAT16(lo, hi);
    }

    const float g = gamma[0];
    float bor[8];
    #pragma unroll
    for (int r = 0; r < 8; ++r) bor[r] = bo[wave * 16 + r + 8 * hh];

    #pragma unroll
    for (int nt = 0; nt < 8; ++nt) {
        v16h b0 = *(const v16h*)&As[nt * 16 + m][16 * hh];
        v16h b1 = *(const v16h*)&As[nt * 16 + m][32 + 16 * hh];
        v8f acc = {};
        acc = wmma_f16(wa[0], b0, acc);
        acc = wmma_f16(wa[1], b1, acc);
        #pragma unroll
        for (int r = 0; r < 8; ++r) {
            int c = wave * 16 + r + 8 * hh;
            size_t off = ((size_t)b * CIN + c) * NTOK + n0 + nt * 16 + m;
            out[off] = g * (acc[r] + bor[r]) + x[off];
        }
    }
}

// =====================================================================
extern "C" void kernel_launch(void* const* d_in, const int* in_sizes, int n_in,
                              void* d_out, int out_size, void* d_ws, size_t ws_size,
                              hipStream_t stream) {
    const float* x     = (const float*)d_in[0];
    const float* wq    = (const float*)d_in[1];
    const float* bq    = (const float*)d_in[2];
    const float* wk    = (const float*)d_in[3];
    const float* bk    = (const float*)d_in[4];
    const float* wv    = (const float*)d_in[5];
    const float* bv    = (const float*)d_in[6];
    const float* wo    = (const float*)d_in[7];
    const float* bo    = (const float*)d_in[8];
    const float* gamma = (const float*)d_in[9];

    char* ws = (char*)d_ws;
    _Float16* Qt  = (_Float16*)(ws);                        // 4 MB
    _Float16* Kt  = (_Float16*)(ws + ((size_t)4  << 20));   // 4 MB
    _Float16* Vn  = (_Float16*)(ws + ((size_t)8  << 20));   // 4 MB
    _Float16* AOh = (_Float16*)(ws + ((size_t)12 << 20));   // 4 MB
    _Float16* Whq = (_Float16*)(ws + ((size_t)16 << 20));
    _Float16* Whk = Whq + 8192;
    _Float16* Whv = Whk + 8192;
    _Float16* Who = Whv + 8192;

    cvt_weights_kernel<<<dim3(128), dim3(256), 0, stream>>>(
        wq, wk, wv, wo, Whq, Whk, Whv, Who);
    qkv_proj_kernel<<<dim3(8 * 32), dim3(256), 0, stream>>>(
        x, Whq, bq, Whk, bk, Whv, bv, Qt, Kt, Vn);
    flash_attn_kernel<<<dim3(8 * 32), dim3(256), 0, stream>>>(
        Qt, Kt, Vn, AOh);
    out_proj_kernel<<<dim3(8 * 32), dim3(256), 0, stream>>>(
        x, Who, bo, gamma, AOh, (float*)d_out);
}
